// Net_hdc_63024350101597
// MI455X (gfx1250) — matrix operand order repaired and checked by reading, not compiled, not run on hardware
//
#include <hip/hip_runtime.h>

typedef __attribute__((ext_vector_type(16))) _Float16 v16h;
typedef __attribute__((ext_vector_type(8)))  _Float16 v8h;
typedef __attribute__((ext_vector_type(8)))  float    v8f;
typedef __attribute__((ext_vector_type(4)))  float    v4f;
typedef __attribute__((ext_vector_type(2)))  float    v2f;

#define THETA     0.999f
#define HID       24
#define SPAD      32            // padded spike-tile width (power of 2, >= 24+8)
#define NSTATES   8
#define NACT      4
#define OPAD      16            // padded logit-tile width
#define WAVES_PB  8
#define THREADS   (WAVES_PB * 32)

// ---- B-fragment builder: B[k][n] = W[n][k], k order used here (same as A):
// element e of v16h <-> K = 16*(lane>>4) + e, column N = (lane&15) + 16*tileN
__device__ inline v16h make_bfrag(const float* __restrict__ W, int fan_in,
                                  int n_out, int tileN, int lane) {
  const int n = (lane & 15) + tileN * 16;
  const int kbase = (lane >> 4) * 16;
  v16h b;
#pragma unroll
  for (int e = 0; e < 16; ++e) {
    const int k = kbase + e;
    const float v = (n < n_out && k < fan_in) ? W[n * fan_in + k] : 0.0f;
    b[e] = (_Float16)v;
  }
  return b;
}

// ---- C-fragment: bias broadcast down each column. C/D layout: N = lane&15.
__device__ inline v8f make_cfrag(const float* __restrict__ bias, int n_out,
                                 int tileN, int lane) {
  const int n = (lane & 15) + tileN * 16;
  const float v = (n < n_out) ? bias[n] : 0.0f;
  v8f c;
#pragma unroll
  for (int r = 0; r < 8; ++r) c[r] = v;
  return c;
}

// ---- spike + store both D fragments into padded 16x32 LDS tile, branch-free.
// D layout: VGPR r <-> M = r + 8*(lane>>4), N = lane&15 (tile1: +16).
// Cols 24..31 receive junk from tile1 lanes n0>=8; never consumed as data.
__device__ inline void spike_to_lds(_Float16* __restrict__ s, v8f t0, v8f t1,
                                    int lane) {
  const int n0 = lane & 15;
  const int mb = (lane >> 4) * 8;
#pragma unroll
  for (int r = 0; r < 8; ++r) {
    const int m = mb + r;
    s[m * SPAD + n0]      = (t0[r] >= THETA) ? (_Float16)1.0f : (_Float16)0.0f;
    s[m * SPAD + 16 + n0] = (t1[r] >= THETA) ? (_Float16)1.0f : (_Float16)0.0f;
  }
}

// ---- rebuild A fragment (16x32 f16 A layout) from padded row-major LDS tile.
// All lanes: two aligned 16B loads (K = 16*half + e); elements 8..15 zero-masked
// for lanes>=16 (they would otherwise alias the junk cols 24..31).
__device__ inline v16h lds_to_afrag(const _Float16* __restrict__ s, int lane) {
  const int m = lane & 15;
  const int half = lane >> 4;                       // 0 or 1
  const _Float16* row = s + m * SPAD;
  const v8h lo = *(const v8h*)(row + half * 16);     // K 0..7  or 16..23
  const v8h hi = *(const v8h*)(row + half * 16 + 8); // K 8..15 or junk
  const bool keep = (half == 0);
  v16h a;
#pragma unroll
  for (int e = 0; e < 8; ++e) {
    a[e]     = lo[e];
    a[8 + e] = keep ? hi[e] : (_Float16)0.0f;        // v_cndmask, no branch
  }
  return a;
}

__global__ __launch_bounds__(THREADS) void snn_fwd_wmma(
    const float* __restrict__ x, const float* __restrict__ noise,
    const float* __restrict__ W1, const float* __restrict__ b1,
    const float* __restrict__ W2, const float* __restrict__ b2,
    const float* __restrict__ W3, const float* __restrict__ b3,
    float* __restrict__ out_s, float* __restrict__ out_hdc, int nTiles) {
  __shared__ __align__(16) _Float16 sstage[WAVES_PB][16 * SPAD]; // spike tiles
  __shared__ __align__(16) float    ostage[WAVES_PB][16 * OPAD]; // logit tiles

  const int lane = threadIdx.x & 31;
  const int wave = threadIdx.x >> 5;
  _Float16* s = sstage[wave];
  float*    os = ostage[wave];

  // ---- hoisted weight/bias fragments (persistent across the tile loop) ----
  const v16h bW1t0 = make_bfrag(W1, NSTATES, HID, 0, lane);
  const v16h bW1t1 = make_bfrag(W1, NSTATES, HID, 1, lane);
  const v16h bW2t0 = make_bfrag(W2, HID, HID, 0, lane);
  const v16h bW2t1 = make_bfrag(W2, HID, HID, 1, lane);
  const v16h bW3   = make_bfrag(W3, HID, NACT, 0, lane);
  const v8f  c1t0 = make_cfrag(b1, HID, 0, lane);
  const v8f  c1t1 = make_cfrag(b1, HID, 1, lane);
  const v8f  c2t0 = make_cfrag(b2, HID, 0, lane);
  const v8f  c2t1 = make_cfrag(b2, HID, 1, lane);
  const v8f  c3   = make_cfrag(b3, NACT, 0, lane);

  const int waveGlobal = blockIdx.x * WAVES_PB + wave;
  const int waveStride = gridDim.x * WAVES_PB;
  const bool loRow = (lane < 16);

  for (int tile = waveGlobal; tile < nTiles; tile += waveStride) {
    const int row0 = tile * 16;

    // prefetch next tile's x (512B tile; lane strides of 16B cover it)
    if (tile + waveStride < nTiles) {
      const char* p = (const char*)(x + (size_t)(tile + waveStride) * 16 * NSTATES);
      __builtin_prefetch(p + lane * 16, 0, 1);
    }

    // ---- layer 1: A = (x + noise) in f16, K=8 real of 32, branch-free ----
    // Lanes 16..31 duplicate lanes 0..15's loads (cache hit) and mask to zero.
    const int rowm = row0 + (lane & 15);
    const float* xr = x + (size_t)rowm * NSTATES;
    const float ns = noise[rowm];
    const v4f x0 = *(const v4f*)(xr);
    const v4f x1 = *(const v4f*)(xr + 4);
    v16h a1;
#pragma unroll
    for (int e = 0; e < 4; ++e) {
      a1[e]      = loRow ? (_Float16)(x0[e] + ns) : (_Float16)0.0f;
      a1[4 + e]  = loRow ? (_Float16)(x1[e] + ns) : (_Float16)0.0f;
      a1[8 + e]  = (_Float16)0.0f;
      a1[12 + e] = (_Float16)0.0f;
    }
    v8f h0 = __builtin_amdgcn_wmma_f32_16x16x32_f16(false, a1, false, bW1t0,
                                                    (short)0, c1t0, false, false);
    v8f h1 = __builtin_amdgcn_wmma_f32_16x16x32_f16(false, a1, false, bW1t1,
                                                    (short)0, c1t1, false, false);
    spike_to_lds(s, h0, h1, lane);
    asm volatile("s_wait_dscnt 0" ::: "memory");  // wave-private LDS RAW

    // ---- layer 2 ----
    const v16h a2 = lds_to_afrag(s, lane);
    v8f g0 = __builtin_amdgcn_wmma_f32_16x16x32_f16(false, a2, false, bW2t0,
                                                    (short)0, c2t0, false, false);
    v8f g1 = __builtin_amdgcn_wmma_f32_16x16x32_f16(false, a2, false, bW2t1,
                                                    (short)0, c2t1, false, false);
    spike_to_lds(s, g0, g1, lane);                // s now holds s2 (= out_hdc)
    asm volatile("s_wait_dscnt 0" ::: "memory");

    // ---- layer 3 ----
    const v16h a3 = lds_to_afrag(s, lane);
    v8f o = __builtin_amdgcn_wmma_f32_16x16x32_f16(false, a3, false, bW3,
                                                   (short)0, c3, false, false);
    // stage logits into padded 16x16 tile, branch-free (cols >=4 are junk)
    {
      const int n0 = lane & 15;
      const int mb = (lane >> 4) * 8;
#pragma unroll
      for (int r = 0; r < 8; ++r) os[(mb + r) * OPAD + n0] = o[r];
    }
    asm volatile("s_wait_dscnt 0" ::: "memory");

    // ---- coalesced global writes ----
    {
      // out_s: 64 floats/tile -> one float2 per lane
      const int orow = lane >> 1;                 // logical row
      const int ocol = (lane & 1) * 2;            // logical col (0 or 2)
      v2f vo;
      vo[0] = os[orow * OPAD + ocol + 0];
      vo[1] = os[orow * OPAD + ocol + 1];
      *(v2f*)(out_s + (size_t)tile * 64 + lane * 2) = vo;

      // out_hdc: 384 floats/tile; lane covers 12 consecutive (row-major over
      // width 24): row = lane/2, col = (lane&1)*12 -> one aligned 24B LDS run
      const int hrow = lane >> 1;
      const int hcol = (lane & 1) * 12;
      const _Float16* src = s + hrow * SPAD + hcol;
      float* hb = out_hdc + (size_t)tile * (16 * HID) + lane * 12;
#pragma unroll
      for (int q = 0; q < 3; ++q) {
        v4f v;
#pragma unroll
        for (int e = 0; e < 4; ++e) v[e] = (float)src[q * 4 + e];
        *(v4f*)(hb + q * 4) = v;
      }
    }
  }
}

extern "C" void kernel_launch(void* const* d_in, const int* in_sizes, int n_in,
                              void* d_out, int out_size, void* d_ws, size_t ws_size,
                              hipStream_t stream) {
  const float* x     = (const float*)d_in[0];
  const float* noise = (const float*)d_in[1];
  const float* W1    = (const float*)d_in[2];
  const float* b1    = (const float*)d_in[3];
  const float* W2    = (const float*)d_in[4];
  const float* b2    = (const float*)d_in[5];
  const float* W3    = (const float*)d_in[6];
  const float* b3    = (const float*)d_in[7];

  const int B = in_sizes[0] / NSTATES;   // 2,097,152
  const int nTiles = B / 16;             // B divisible by 16

  float* out_s   = (float*)d_out;                      // [B,4]
  float* out_hdc = (float*)d_out + (size_t)B * NACT;   // [B,24,1]

  int blocks = 1024;                     // persistent-ish: ~16 tiles per wave
  const int maxBlocks = (nTiles + WAVES_PB - 1) / WAVES_PB;
  if (blocks > maxBlocks) blocks = maxBlocks;

  snn_fwd_wmma<<<blocks, THREADS, 0, stream>>>(x, noise, W1, b1, W2, b2, W3, b3,
                                               out_s, out_hdc, nTiles);
}
